// Decoder_58566174048778
// MI455X (gfx1250) — matrix operand order repaired and checked by reading, not compiled, not run on hardware
//
#include <hip/hip_runtime.h>
#include <math.h>

// ---------------- problem constants (match reference) ----------------
#define Bk   16
#define Lk   1024
#define Hk   512
#define Dk   768
#define Vk   30522
#define Tk   64
#define PADT 0
#define CLST 101
#define MASKT 103
#define NEGV (-1000000.0f)
#define LNEPS 1e-12f

// ---------------- types ----------------
typedef __bf16 bf16;
typedef __attribute__((ext_vector_type(16))) __bf16 v16bf;
typedef __attribute__((ext_vector_type(8)))  __bf16 v8bf;
typedef __attribute__((ext_vector_type(8)))  float  v8f;

union AFrag { v16bf v; v8bf h[2]; };

__device__ __forceinline__ bf16 f2b(float f) {
  unsigned u = __float_as_uint(f);
  unsigned r = u + 0x7fffu + ((u >> 16) & 1u);      // round-to-nearest-even
  unsigned short s = (unsigned short)(r >> 16);
  bf16 b; __builtin_memcpy(&b, &s, 2); return b;
}
__device__ __forceinline__ float sigmoidf(float x) { return 1.f / (1.f + expf(-x)); }

// block reductions, blockDim.x == 256
__device__ __forceinline__ float blockSum(float v, float* red) {
  int tid = threadIdx.x;
  red[tid] = v; __syncthreads();
  for (int s = 128; s > 0; s >>= 1) { if (tid < s) red[tid] += red[tid + s]; __syncthreads(); }
  float r = red[0]; __syncthreads(); return r;
}
__device__ __forceinline__ float blockMax(float v, float* red) {
  int tid = threadIdx.x;
  red[tid] = v; __syncthreads();
  for (int s = 128; s > 0; s >>= 1) { if (tid < s) red[tid] = fmaxf(red[tid], red[tid + s]); __syncthreads(); }
  float r = red[0]; __syncthreads(); return r;
}

// ---------------- WMMA GEMM: out[16,N] = A[16,K](bf16) * W[N,K](bf16)^T + bias ----------------
// one wave per 16x16 output tile; K multiple of 32; N multiple of 16.
#define GEMM_WAVES 4
__global__ void k_gemm16(const bf16* __restrict__ A, const bf16* __restrict__ W,
                         const float* __restrict__ bias, float* __restrict__ outF,
                         bf16* __restrict__ outB, int bstride, int K, int N) {
  int wave = blockIdx.x * GEMM_WAVES + (threadIdx.x >> 5);
  int lane = threadIdx.x & 31;
  int half = lane >> 4;
  int mn   = lane & 15;
  int n0   = wave * 16;
  if (n0 >= N) return;                              // wave-uniform: EXEC stays all-1s in live waves

  v8f acc;
#pragma unroll
  for (int i = 0; i < 8; ++i) acc[i] = 0.f;

  const bf16* arow = A + (size_t)mn * K + half * 16; // A: lane row mn, 16 contiguous k
  const bf16* wrow = W + (size_t)(n0 + mn) * K + half * 16; // B: row n, 16 contiguous k

  for (int k0 = 0; k0 < K; k0 += 32) {
    AFrag a;
    a.h[0] = *(const v8bf*)(arow + k0);             // K = k0 + half*16 + [0..7]
    a.h[1] = *(const v8bf*)(arow + k0 + 8);         // K = k0 + half*16 + [8..15]
    v16bf bb = *(const v16bf*)(wrow + k0);          // K = k0 + half*16 + [0..15]
    acc = __builtin_amdgcn_wmma_f32_16x16x32_bf16(false, a.v, false, bb,
                                                  (short)0, acc, false, false);
  }
  int n = n0 + mn;
  float bv = bias ? bias[n] : 0.f;
#pragma unroll
  for (int v = 0; v < 8; ++v) {
    int m = v + half * 8;                           // C/D layout: M = vgpr + half*8
    float val = acc[v] + bv;
    outF[(size_t)m * N + n] = val;
    if (outB) outB[(size_t)m * bstride + n] = f2b(val);
  }
}

// ---------------- prologue kernels ----------------
__global__ void k_f32_to_bf16(const float* __restrict__ src, bf16* __restrict__ dst, int n) {
  int i = blockIdx.x * 256 + threadIdx.x;
  if (i < n) dst[i] = f2b(src[i]);
}
__global__ void k_zero_f32(float* p, int n) {
  int i = blockIdx.x * 256 + threadIdx.x; if (i < n) p[i] = 0.f;
}
__global__ void k_zero_bf16(bf16* p, int n) {
  int i = blockIdx.x * 256 + threadIdx.x; if (i < n) p[i] = f2b(0.f);
}
__global__ void k_miss_init(unsigned char* miss, int n) {
  int i = blockIdx.x * 256 + threadIdx.x; if (i < n) miss[i] = 1;
}
__global__ void k_miss_scatter(const int* __restrict__ io, const int* __restrict__ ic,
                               unsigned char* miss) {
  int i = blockIdx.x * 256 + threadIdx.x;
  if (i >= Bk * 2 * Lk) return;
  int b = i / (2 * Lk), l = i % (2 * Lk);
  int id = (l < Lk) ? io[b * Lk + l] : ic[b * Lk + (l - Lk)];
  miss[(size_t)b * Vk + id] = 0;                    // present in either sequence -> not missing
}
__global__ void k_out0(float* dec_out, float* samp_out, int* sampled) {
  int i = blockIdx.x * 256 + threadIdx.x;
  if (i < Bk * Vk) {
    int b = i / Vk, v = i % Vk;
    dec_out[(size_t)b * Tk * Vk + v] = (v == CLST) ? 1.f : 0.f;   // SOS one-hot at t=0
  }
  if (i < Bk) { samp_out[i * Tk] = (float)CLST; sampled[i] = CLST; }
}

// ---------------- per-step kernels ----------------
// attention: scores over L for old/change, softmax, contexts -> bf16 into x[:, 0:H] / [H:2H]
__global__ void k_attn(const float* __restrict__ old_, const float* __restrict__ chg,
                       const float* __restrict__ th, bf16* __restrict__ x) {
  __shared__ float th_s[Hk];
  __shared__ float sco[Lk];
  __shared__ float scc[Lk];
  __shared__ float red[256];
  int b = blockIdx.x, tid = threadIdx.x;
  for (int h = tid; h < Hk; h += 256) th_s[h] = th[b * Hk + h];
  __syncthreads();
  for (int l = tid; l < Lk; l += 256) {
    const float* ro = old_ + ((size_t)b * Lk + l) * Hk;
    const float* rc = chg  + ((size_t)b * Lk + l) * Hk;
    float so = 0.f, sc = 0.f;
    for (int h = 0; h < Hk; h += 4) {
      float4 o4 = *(const float4*)(ro + h);
      float4 c4 = *(const float4*)(rc + h);
      so += o4.x * th_s[h] + o4.y * th_s[h + 1] + o4.z * th_s[h + 2] + o4.w * th_s[h + 3];
      sc += c4.x * th_s[h] + c4.y * th_s[h + 1] + c4.z * th_s[h + 2] + c4.w * th_s[h + 3];
    }
    sco[l] = so; scc[l] = sc;
  }
  __syncthreads();
  // softmax over L (each matrix)
  float mo = -3.4e38f, mc = -3.4e38f;
  for (int l = tid; l < Lk; l += 256) { mo = fmaxf(mo, sco[l]); mc = fmaxf(mc, scc[l]); }
  mo = blockMax(mo, red); mc = blockMax(mc, red);
  float so = 0.f, sc = 0.f;
  for (int l = tid; l < Lk; l += 256) {
    float eo = expf(sco[l] - mo); sco[l] = eo; so += eo;
    float ec = expf(scc[l] - mc); scc[l] = ec; sc += ec;
  }
  so = blockSum(so, red); sc = blockSum(sc, red);
  float invo = 1.f / so, invc = 1.f / sc;
  for (int l = tid; l < Lk; l += 256) { sco[l] *= invo; scc[l] *= invc; }
  __syncthreads();
  // contexts
  for (int h = tid; h < Hk; h += 256) {
    float ao = 0.f, ac = 0.f;
    for (int l = 0; l < Lk; ++l) {
      ao += sco[l] * old_[((size_t)b * Lk + l) * Hk + h];
      ac += scc[l] * chg [((size_t)b * Lk + l) * Hk + h];
    }
    x[(size_t)b * 4 * Hk + h]      = f2b(ao);
    x[(size_t)b * 4 * Hk + Hk + h] = f2b(ac);
  }
}

// embed + layernorm -> bf16 [B,D]
__global__ void k_embed(const int* __restrict__ targets, int t,
                        const float* __restrict__ wemb, const float* __restrict__ pemb,
                        const float* __restrict__ temb, const float* __restrict__ g,
                        const float* __restrict__ beta, bf16* __restrict__ eln) {
  __shared__ float e_s[Dk];
  __shared__ float red[256];
  int b = blockIdx.x, tid = threadIdx.x;
  int tok = targets[b * Tk + (t - 1)];
  float lsum = 0.f;
  for (int i = tid; i < Dk; i += 256) {
    float e = wemb[(size_t)tok * Dk + i] + pemb[i] + temb[i];
    e_s[i] = e; lsum += e;
  }
  float mu = blockSum(lsum, red) / Dk;
  float lv = 0.f;
  for (int i = tid; i < Dk; i += 256) { float d = e_s[i] - mu; lv += d * d; }
  float var = blockSum(lv, red) / Dk;
  float inv = rsqrtf(var + LNEPS);
  for (int i = tid; i < Dk; i += 256)
    eln[(size_t)b * Dk + i] = f2b((e_s[i] - mu) * inv * g[i] + beta[i]);
}

// GRU elementwise (torch gate order r,z,n)
__global__ void k_gru(const float* __restrict__ gi, const float* __restrict__ gh,
                      float* __restrict__ hidden, bf16* __restrict__ hidden_bf) {
  int i = blockIdx.x * 256 + threadIdx.x;
  if (i >= Bk * Hk) return;
  int b = i / Hk, h = i % Hk;
  const float* gib = gi + (size_t)b * 3 * Hk;
  const float* ghb = gh + (size_t)b * 3 * Hk;
  float r = sigmoidf(gib[h] + ghb[h]);
  float z = sigmoidf(gib[Hk + h] + ghb[Hk + h]);
  float n = tanhf(gib[2 * Hk + h] + r * ghb[2 * Hk + h]);
  float h2 = (1.f - z) * n + z * hidden[i];
  hidden[i] = h2; hidden_bf[i] = f2b(h2);
}

// per-position copy scores over concatenated encoder: cs[b, 0:2L]
__global__ void k_copyscores(const float* __restrict__ old_, const float* __restrict__ chg,
                             const float* __restrict__ ch, float* __restrict__ cs) {
  __shared__ float ch_s[Hk];
  int b = blockIdx.x, tid = threadIdx.x;
  for (int h = tid; h < Hk; h += 256) ch_s[h] = ch[b * Hk + h];
  __syncthreads();
  for (int l = tid; l < 2 * Lk; l += 256) {
    const float* row = (l < Lk) ? old_ + ((size_t)b * Lk + l) * Hk
                                : chg  + ((size_t)b * Lk + (l - Lk)) * Hk;
    float s = 0.f;
    for (int h = 0; h < Hk; h += 4) {
      float4 r4 = *(const float4*)(row + h);
      s += r4.x * ch_s[h] + r4.y * ch_s[h + 1] + r4.z * ch_s[h + 2] + r4.w * ch_s[h + 3];
    }
    cs[(size_t)b * 2 * Lk + l] = s;
  }
}

// sc init: masked (missing from both, or MASK/PAD) -> 2*NEG (NEG from each branch), else 0
__global__ void k_sc_init(const unsigned char* __restrict__ miss, float* __restrict__ sc) {
  int i = blockIdx.x * 256 + threadIdx.x;
  if (i >= Bk * Vk) return;
  int v = i % Vk;
  bool m = miss[i] || v == MASKT || v == PADT;
  sc[i] = m ? 2.f * NEGV : 0.f;
}
// scatter copy scores to vocab; skip masked ids (reference replaces them with NEG)
__global__ void k_sc_scatter(const int* __restrict__ io, const int* __restrict__ ic,
                             const float* __restrict__ cs, const unsigned char* __restrict__ miss,
                             float* __restrict__ sc) {
  int i = blockIdx.x * 256 + threadIdx.x;
  if (i >= Bk * 2 * Lk) return;
  int b = i / (2 * Lk), l = i % (2 * Lk);
  int id = (l < Lk) ? io[b * Lk + l] : ic[b * Lk + (l - Lk)];
  if (miss[(size_t)b * Vk + id] || id == MASKT || id == PADT) return;
  atomicAdd(&sc[(size_t)b * Vk + id], cs[i]);
}

// softmax over V, write log(softmax+1e-10) into dec_out[b,t,:], argmax -> sampled
__global__ void k_softmaxv(const float* __restrict__ sc, float* __restrict__ dec_out,
                           float* __restrict__ samp_out, int* __restrict__ sampled, int t) {
  __shared__ float redv[256];
  __shared__ int   redi[256];
  __shared__ float red[256];
  int b = blockIdx.x, tid = threadIdx.x;
  const float* row = sc + (size_t)b * Vk;
  float bm = -3.4e38f; int bi = 0;
  for (int v = tid; v < Vk; v += 256) { float x = row[v]; if (x > bm) { bm = x; bi = v; } }
  redv[tid] = bm; redi[tid] = bi; __syncthreads();
  for (int s = 128; s > 0; s >>= 1) {
    if (tid < s) {
      if (redv[tid + s] > redv[tid] ||
          (redv[tid + s] == redv[tid] && redi[tid + s] < redi[tid])) {
        redv[tid] = redv[tid + s]; redi[tid] = redi[tid + s];
      }
    }
    __syncthreads();
  }
  float mx = redv[0]; int amax = redi[0]; __syncthreads();
  float sum = 0.f;
  for (int v = tid; v < Vk; v += 256) sum += expf(row[v] - mx);
  sum = blockSum(sum, red);
  float inv = 1.f / sum;
  float* out = dec_out + ((size_t)b * Tk + t) * Vk;
  for (int v = tid; v < Vk; v += 256) out[v] = logf(expf(row[v] - mx) * inv + 1e-10f);
  if (tid == 0) { sampled[b] = amax; samp_out[b * Tk + t] = (float)amax; }
}

// selective read over concatenated encoder; writes bf16 into x[:, 2H:3H] for next step
__global__ void k_selread(const float* __restrict__ old_, const float* __restrict__ chg,
                          const int* __restrict__ io, const int* __restrict__ ic,
                          const float* __restrict__ cs, const int* __restrict__ sampled,
                          bf16* __restrict__ x) {
  __shared__ float ss[2 * Lk];
  __shared__ float red[256];
  int b = blockIdx.x, tid = threadIdx.x;
  int sm = sampled[b];
  float asum = 0.f;
  for (int l = tid; l < 2 * Lk; l += 256) {
    int id = (l < Lk) ? io[b * Lk + l] : ic[b * Lk + (l - Lk)];
    float s = (id == sm) ? cs[(size_t)b * 2 * Lk + l] : 0.f;
    ss[l] = s; asum += fabsf(s);
  }
  float norm = fmaxf(blockSum(asum, red), 1e-12f);   // blockSum syncs -> ss visible
  float inv = 1.f / norm;
  for (int h = tid; h < Hk; h += 256) {
    float acc = 0.f;
    for (int l = 0; l < 2 * Lk; ++l) {
      float s = ss[l];
      if (s != 0.f) {
        const float* row = (l < Lk) ? old_ + ((size_t)b * Lk + l) * Hk
                                    : chg  + ((size_t)b * Lk + (l - Lk)) * Hk;
        acc += s * row[h];
      }
    }
    x[(size_t)b * 4 * Hk + 2 * Hk + h] = f2b(acc * inv);
  }
}

// ---------------- host ----------------
static inline int grid256(int n) { return (n + 255) / 256; }

extern "C" void kernel_launch(void* const* d_in, const int* in_sizes, int n_in,
                              void* d_out, int out_size, void* d_ws, size_t ws_size,
                              hipStream_t stream) {
  // inputs: setup_inputs() dict order, params flattened in make_params insertion order
  const float* old_   = (const float*)d_in[0];   // [B,L,H]
  const float* chg    = (const float*)d_in[1];   // [B,L,H]
  const int*   io     = (const int*)  d_in[2];   // [B,L]
  const int*   ic     = (const int*)  d_in[3];   // [B,L]
  const int*   tgt    = (const int*)  d_in[4];   // [B,T]
  const float* wemb   = (const float*)d_in[5];   // [V,D]
  const float* pemb   = (const float*)d_in[6];   // [P,D]
  const float* temb   = (const float*)d_in[7];   // [2,D]
  const float* ln_g   = (const float*)d_in[8];
  const float* ln_b   = (const float*)d_in[9];
  const float* ffW    = (const float*)d_in[10];  // [H,D]
  const float* ffb    = (const float*)d_in[11];
  const float* attnW  = (const float*)d_in[12];  // [H,H]
  const float* attnb  = (const float*)d_in[13];
  const float* copyW  = (const float*)d_in[14];  // [H,H]
  const float* copyb  = (const float*)d_in[15];
  const float* Wih    = (const float*)d_in[16];  // [3H,4H]
  const float* Whh    = (const float*)d_in[17];  // [3H,H]
  const float* bih    = (const float*)d_in[18];
  const float* bhh    = (const float*)d_in[19];

  float* dec_out  = (float*)d_out;                               // [B,T,V]
  float* samp_out = (float*)d_out + (size_t)Bk * Tk * Vk;        // [B,T]

  // workspace carve (256B aligned)
  size_t off = 0;
  auto carve = [&](size_t bytes) -> char* {
    char* p = (char*)d_ws + off;
    off = (off + bytes + 255) & ~(size_t)255;
    return p;
  };
  bf16* attnW_b = (bf16*)carve((size_t)Hk * Hk * 2);
  bf16* copyW_b = (bf16*)carve((size_t)Hk * Hk * 2);
  bf16* ffW_b   = (bf16*)carve((size_t)Hk * Dk * 2);
  bf16* Wih_b   = (bf16*)carve((size_t)3 * Hk * 4 * Hk * 2);
  bf16* Whh_b   = (bf16*)carve((size_t)3 * Hk * Hk * 2);
  bf16* x_b     = (bf16*)carve((size_t)Bk * 4 * Hk * 2);
  bf16* hid_b   = (bf16*)carve((size_t)Bk * Hk * 2);
  bf16* eln_b   = (bf16*)carve((size_t)Bk * Dk * 2);
  float* th     = (float*)carve((size_t)Bk * Hk * 4);   // reused for emb-f32 and ch
  float* gi     = (float*)carve((size_t)Bk * 3 * Hk * 4);
  float* gh     = (float*)carve((size_t)Bk * 3 * Hk * 4);
  float* cs     = (float*)carve((size_t)Bk * 2 * Lk * 4);
  float* sc     = (float*)carve((size_t)Bk * Vk * 4);
  float* hid_f  = (float*)carve((size_t)Bk * Hk * 4);
  unsigned char* miss = (unsigned char*)carve((size_t)Bk * Vk);
  int* sampled  = (int*)carve(Bk * 4);
  (void)ws_size; (void)n_in; (void)in_sizes; (void)out_size;

  // ---- prologue: weight conversion, miss mask, state init, t=0 outputs ----
  k_f32_to_bf16<<<grid256(Hk * Hk), 256, 0, stream>>>(attnW, attnW_b, Hk * Hk);
  k_f32_to_bf16<<<grid256(Hk * Hk), 256, 0, stream>>>(copyW, copyW_b, Hk * Hk);
  k_f32_to_bf16<<<grid256(Hk * Dk), 256, 0, stream>>>(ffW, ffW_b, Hk * Dk);
  k_f32_to_bf16<<<grid256(3 * Hk * 4 * Hk), 256, 0, stream>>>(Wih, Wih_b, 3 * Hk * 4 * Hk);
  k_f32_to_bf16<<<grid256(3 * Hk * Hk), 256, 0, stream>>>(Whh, Whh_b, 3 * Hk * Hk);
  k_miss_init<<<grid256(Bk * Vk), 256, 0, stream>>>(miss, Bk * Vk);
  k_miss_scatter<<<grid256(Bk * 2 * Lk), 256, 0, stream>>>(io, ic, miss);
  k_zero_f32<<<grid256(Bk * Hk), 256, 0, stream>>>(hid_f, Bk * Hk);
  k_zero_bf16<<<grid256(Bk * Hk), 256, 0, stream>>>(hid_b, Bk * Hk);
  k_zero_bf16<<<grid256(Bk * 4 * Hk), 256, 0, stream>>>(x_b, Bk * 4 * Hk); // sel slot = 0
  k_out0<<<grid256(Bk * Vk), 256, 0, stream>>>(dec_out, samp_out, sampled);

  const int gN512  = (Hk / 16 + GEMM_WAVES - 1) / GEMM_WAVES;          // 8 blocks
  const int gN1536 = (3 * Hk / 16 + GEMM_WAVES - 1) / GEMM_WAVES;      // 24 blocks

  // ---- sequential decode, t = 1 .. T-1 ----
  for (int t = 1; t < Tk; ++t) {
    // th = hidden @ attn_W^T + attn_b
    k_gemm16<<<gN512, 32 * GEMM_WAVES, 0, stream>>>(hid_b, attnW_b, attnb, th,
                                                    (bf16*)nullptr, 0, Hk, Hk);
    // attention + contexts -> x[:,0:2H]
    k_attn<<<Bk, 256, 0, stream>>>(old_, chg, th, x_b);
    // embed + LN
    k_embed<<<Bk, 256, 0, stream>>>(tgt, t, wemb, pemb, temb, ln_g, ln_b, eln_b);
    // emb = eln @ ff_W^T + ff_b  (bf16 result straight into x[:,3H:4H])
    k_gemm16<<<gN512, 32 * GEMM_WAVES, 0, stream>>>(eln_b, ffW_b, ffb, th,
                                                    x_b + 3 * Hk, 4 * Hk, Dk, Hk);
    // gi = x @ Wih^T + bih ; gh = hidden @ Whh^T + bhh
    k_gemm16<<<gN1536, 32 * GEMM_WAVES, 0, stream>>>(x_b, Wih_b, bih, gi,
                                                     (bf16*)nullptr, 0, 4 * Hk, 3 * Hk);
    k_gemm16<<<gN1536, 32 * GEMM_WAVES, 0, stream>>>(hid_b, Whh_b, bhh, gh,
                                                     (bf16*)nullptr, 0, Hk, 3 * Hk);
    // GRU cell -> new hidden (f32 + bf16)
    k_gru<<<grid256(Bk * Hk), 256, 0, stream>>>(gi, gh, hid_f, hid_b);
    // ch = h2 @ copy_W^T + copy_b   (reuse th buffer)
    k_gemm16<<<gN512, 32 * GEMM_WAVES, 0, stream>>>(hid_b, copyW_b, copyb, th,
                                                    (bf16*)nullptr, 0, Hk, Hk);
    // per-position copy scores over old||change
    k_copyscores<<<Bk, 256, 0, stream>>>(old_, chg, th, cs);
    // scatter to vocab with miss masking, softmax over V, argmax, log-probs out
    k_sc_init<<<grid256(Bk * Vk), 256, 0, stream>>>(miss, sc);
    k_sc_scatter<<<grid256(Bk * 2 * Lk), 256, 0, stream>>>(io, ic, cs, miss, sc);
    k_softmaxv<<<Bk, 256, 0, stream>>>(sc, dec_out, samp_out, sampled, t);
    // selective read -> x[:,2H:3H] for next step
    k_selread<<<Bk, 256, 0, stream>>>(old_, chg, io, ic, cs, sampled, x_b);
  }
}
